// AttnGraphPooling_87196426043574
// MI455X (gfx1250) — compile-verified, matrix operand order checked by reading, not run on hardware
//
#include <hip/hip_runtime.h>
#include <hip/hip_bf16.h>
#include <math.h>

#define D_MODEL 256

typedef __bf16 v16bf  __attribute__((ext_vector_type(16)));
typedef __bf16 bf16x2 __attribute__((ext_vector_type(2)));
typedef float  v8f    __attribute__((ext_vector_type(8)));

union BFrag {
    v16bf  v;
    bf16x2 h[8];
    unsigned int w[8];
    uint4  q[2];
};

// packed fp32 -> bf16 (RNE). Hardware v_cvt_pk_bf16_f32 when available.
__device__ __forceinline__ bf16x2 cvt_pk_bf16(float lo, float hi) {
#if __has_builtin(__builtin_amdgcn_cvt_pk_bf16_f32)
    return __builtin_amdgcn_cvt_pk_bf16_f32(lo, hi);
#else
    bf16x2 r;
    r.x = (__bf16)lo;
    r.y = (__bf16)hi;
    return r;
#endif
}

__device__ __forceinline__ unsigned short f2bf(float x) {
    unsigned int b = __float_as_uint(x);
    b += 0x7FFFu + ((b >> 16) & 1u);
    return (unsigned short)(b >> 16);
}

// float atomic-max via integer atomics (valid with -inf init):
// non-negative values: signed int max; negative values: unsigned min.
__device__ __forceinline__ void atomicMaxFloat(float* addr, float v) {
    if (v >= 0.0f)
        atomicMax((int*)addr, __float_as_int(v));
    else
        atomicMin((unsigned int*)addr, (unsigned int)__float_as_int(v));
}

__global__ __launch_bounds__(256) void init_kernel(float* seg_max, float* seg_sum,
                                                   float* outf, int n) {
    int i = blockIdx.x * blockDim.x + threadIdx.x;
    if (i < n) {
        seg_max[i] = -INFINITY;
        seg_sum[i] = 0.0f;
        outf[i]    = 0.0f;
    }
}

__global__ __launch_bounds__(256) void cvt_weights(const float* __restrict__ kW,
                                                   const float* __restrict__ vW,
                                                   unsigned short* __restrict__ kB,
                                                   unsigned short* __restrict__ vB, int n) {
    int i = blockIdx.x * blockDim.x + threadIdx.x;
    if (i < n) {
        kB[i] = f2bf(kW[i]);
        vB[i] = f2bf(vW[i]);
    }
}

// PASS 1: attn GEMM -> atomic segment max
// PASS 2: attn GEMM -> exp(attn-max) -> atomic segment sum
// PASS 3: attn + value GEMMs -> atomic add  w * val  into f_graph (d_out)
template <int PASS>
__global__ __launch_bounds__(256) void attn_pass(
    const float* __restrict__ f_node,
    const unsigned short* __restrict__ kW,      // bf16 [256,256] row-major (out,in)
    const unsigned short* __restrict__ vW,      // bf16 [256,256]
    const float* __restrict__ key_b,
    const float* __restrict__ value_b,
    const int* __restrict__ gid,
    float* __restrict__ seg_max,
    float* __restrict__ seg_sum,
    float* __restrict__ fgraph,
    int L)
{
    const int lane = threadIdx.x & 31;
    const int wave = threadIdx.x >> 5;
    const int lrow = lane & 15;     // N index for B/C, M index for A
    const int lgrp = lane >> 4;     // K-group for A/B, +8 rows for C

    const int row_base = blockIdx.x * 32 + (wave >> 2) * 16;
    const int col_base = (wave & 3) * 64;

    v8f accK[4] = {};
    v8f accV[4] = {};

    int arow = row_base + lrow;
    if (arow >= L) arow = L - 1;
    const float* aptr = f_node + (size_t)arow * D_MODEL;

#pragma unroll
    for (int ks = 0; ks < 8; ++ks) {
        const int k0 = ks * 32 + lgrp * 8;
        // speculative prefetch of next K chunk -> global_prefetch_b8
        __builtin_prefetch(aptr + k0 + 32, 0, 3);

        // A fragment: K = {k0..k0+7, k0+16..k0+23}, fp32 -> bf16 via v_cvt_pk
        float4 f0 = *(const float4*)(aptr + k0);
        float4 f1 = *(const float4*)(aptr + k0 + 4);
        float4 f2 = *(const float4*)(aptr + k0 + 16);
        float4 f3 = *(const float4*)(aptr + k0 + 20);
        BFrag a;
        a.h[0] = cvt_pk_bf16(f0.x, f0.y); a.h[1] = cvt_pk_bf16(f0.z, f0.w);
        a.h[2] = cvt_pk_bf16(f1.x, f1.y); a.h[3] = cvt_pk_bf16(f1.z, f1.w);
        a.h[4] = cvt_pk_bf16(f2.x, f2.y); a.h[5] = cvt_pk_bf16(f2.z, f2.w);
        a.h[6] = cvt_pk_bf16(f3.x, f3.y); a.h[7] = cvt_pk_bf16(f3.z, f3.w);

        // hoist ALL B fragments first so loads batch into clauses and
        // overlap with the WMMA group below
        BFrag bk[4];
#pragma unroll
        for (int t = 0; t < 4; ++t) {
            const unsigned short* p = kW + (size_t)(col_base + t * 16 + lrow) * D_MODEL + k0;
            bk[t].q[0] = *(const uint4*)(p);
            bk[t].q[1] = *(const uint4*)(p + 16);
        }
        BFrag bv[4];
        if (PASS == 3) {
#pragma unroll
            for (int t = 0; t < 4; ++t) {
                const unsigned short* p = vW + (size_t)(col_base + t * 16 + lrow) * D_MODEL + k0;
                bv[t].q[0] = *(const uint4*)(p);
                bv[t].q[1] = *(const uint4*)(p + 16);
            }
        }

#pragma unroll
        for (int t = 0; t < 4; ++t) {
            accK[t] = __builtin_amdgcn_wmma_f32_16x16x32_bf16(
                false, a.v, false, bk[t].v, (short)0, accK[t], false, false);
            if (PASS == 3) {
                accV[t] = __builtin_amdgcn_wmma_f32_16x16x32_bf16(
                    false, a.v, false, bv[t].v, (short)0, accV[t], false, false);
            }
        }
    }

    // Per-row graph ids for this lane's 8 C-rows (hoisted out of tile loop)
    int g8[8];
    bool ok8[8];
#pragma unroll
    for (int i = 0; i < 8; ++i) {
        int row = row_base + i + lgrp * 8;
        ok8[i] = (row < L);
        g8[i] = gid[ok8[i] ? row : (L - 1)];
    }

    // Epilogue: C/D layout -> col = col_base + t*16 + (lane&15), row = i + 8*(lane>>4)
#pragma unroll
    for (int t = 0; t < 4; ++t) {
        const int col = col_base + t * 16 + lrow;
        const float kb = key_b[col];
        const float vb = (PASS == 3) ? value_b[col] : 0.0f;
#pragma unroll
        for (int i = 0; i < 8; ++i) {
            if (!ok8[i]) continue;
            const size_t o = (size_t)g8[i] * D_MODEL + col;
            const float attn = accK[t][i] + kb;
            if (PASS == 1) {
                atomicMaxFloat(seg_max + o, attn);
            } else if (PASS == 2) {
                const float ex = __expf(attn - seg_max[o]);
                unsafeAtomicAdd(seg_sum + o, ex);
            } else {
                const float w = __expf(attn - seg_max[o]) / (seg_sum[o] + 1e-7f);
                const float val = accV[t][i] + vb;
                unsafeAtomicAdd(fgraph + o, w * val);
            }
        }
    }
}

// In-place LayerNorm over rows of [G, 256]; one block per row, 8 wave32 per block.
__global__ __launch_bounds__(256) void ln_kernel(float* __restrict__ data,
                                                 const float* __restrict__ gamma,
                                                 const float* __restrict__ beta) {
    __shared__ float smem[8];
    const int row = blockIdx.x;
    const int tid = threadIdx.x;
    float v = data[(size_t)row * D_MODEL + tid];

    float s = v;
#pragma unroll
    for (int o = 16; o > 0; o >>= 1) s += __shfl_xor(s, o, 32);
    if ((tid & 31) == 0) smem[tid >> 5] = s;
    __syncthreads();
    float tot = 0.0f;
#pragma unroll
    for (int i = 0; i < 8; ++i) tot += smem[i];
    const float mu = tot * (1.0f / 256.0f);
    __syncthreads();

    const float d = v - mu;
    float sq = d * d;
#pragma unroll
    for (int o = 16; o > 0; o >>= 1) sq += __shfl_xor(sq, o, 32);
    if ((tid & 31) == 0) smem[tid >> 5] = sq;
    __syncthreads();
    float vtot = 0.0f;
#pragma unroll
    for (int i = 0; i < 8; ++i) vtot += smem[i];
    const float var = vtot * (1.0f / 256.0f);

    data[(size_t)row * D_MODEL + tid] = d * rsqrtf(var + 1e-5f) * gamma[tid] + beta[tid];
}

extern "C" void kernel_launch(void* const* d_in, const int* in_sizes, int n_in,
                              void* d_out, int out_size, void* d_ws, size_t ws_size,
                              hipStream_t stream) {
    const float* f_node  = (const float*)d_in[0];
    const float* key_W   = (const float*)d_in[1];
    const float* key_b   = (const float*)d_in[2];
    const float* value_W = (const float*)d_in[3];
    const float* value_b = (const float*)d_in[4];
    const float* gamma   = (const float*)d_in[5];
    const float* beta    = (const float*)d_in[6];
    const int*   gid     = (const int*)d_in[7];

    const int L  = in_sizes[0] / D_MODEL;   // 500000
    const int GD = out_size;                // G * 256

    // workspace layout
    char* ws = (char*)d_ws;
    unsigned short* kB = (unsigned short*)ws;                        // 256*256 bf16
    unsigned short* vB = kB + D_MODEL * D_MODEL;                     // 256*256 bf16
    float* seg_max = (float*)(ws + 2 * (size_t)D_MODEL * D_MODEL * sizeof(unsigned short));
    float* seg_sum = seg_max + GD;
    float* outf    = (float*)d_out;                                  // f_graph accumulator

    init_kernel<<<(GD + 255) / 256, 256, 0, stream>>>(seg_max, seg_sum, outf, GD);
    cvt_weights<<<(D_MODEL * D_MODEL + 255) / 256, 256, 0, stream>>>(
        key_W, value_W, kB, vB, D_MODEL * D_MODEL);

    const int blocks = (L + 31) / 32;
    attn_pass<1><<<blocks, 256, 0, stream>>>(f_node, kB, vB, key_b, value_b, gid,
                                             seg_max, seg_sum, outf, L);
    attn_pass<2><<<blocks, 256, 0, stream>>>(f_node, kB, vB, key_b, value_b, gid,
                                             seg_max, seg_sum, outf, L);
    attn_pass<3><<<blocks, 256, 0, stream>>>(f_node, kB, vB, key_b, value_b, gid,
                                             seg_max, seg_sum, outf, L);

    ln_kernel<<<GD / D_MODEL, 256, 0, stream>>>(outf, gamma, beta);
}
